// SemsegCDRLink_48284022341777
// MI455X (gfx1250) — matrix operand order repaired and checked by reading, not compiled, not run on hardware
//
#include <hip/hip_runtime.h>
#include <cstdint>
#include <cstddef>

// Problem constants from the reference
#define K_NBR  27
#define N_VOX  200000
#define V_NUM  3
#define B_NUM  4
#define C_CH   64
#define H_IMG  120
#define W_IMG  160
#define D3_CH  96
#define EPSV   1e-5f

typedef __bf16 v16bf __attribute__((ext_vector_type(16)));
typedef float  v8f   __attribute__((ext_vector_type(8)));

union Frag16 { v16bf v; uint4 q[2]; unsigned u[8]; };

__device__ __forceinline__ unsigned short f2bf(float f) {
    unsigned u = __float_as_uint(f);
    u += 0x7FFFu + ((u >> 16) & 1u);     // round-to-nearest-even
    return (unsigned short)(u >> 16);
}

// ---------------------------------------------------------------------------
// Merge in_idx + nbr_mask into a single index array (-1 == masked out)
// ---------------------------------------------------------------------------
__global__ void prep_idx_kernel(const int* __restrict__ in_idx,
                                const unsigned char* __restrict__ mask,
                                int* __restrict__ idx2, int total) {
    int e = blockIdx.x * blockDim.x + threadIdx.x;
    if (e < total) idx2[e] = mask[e] ? in_idx[e] : -1;
}

// ---------------------------------------------------------------------------
// Pack f32 weights (K, Cin, Cout) into bf16 WMMA B-fragment order.
// Fragment (k, ct, cot): 32 lanes x 8 dwords. lane holds col = cot*16+(lane&15);
// lanes<16 hold K {0..7,16..23}, lanes>=16 K {8..15,24..31}; dword j packs (K, K+1).
// ---------------------------------------------------------------------------
__global__ void pack_w_kernel(const float* __restrict__ W, uint32_t* __restrict__ Wp,
                              int Cin, int Cout, int total) {
    int tid = blockIdx.x * blockDim.x + threadIdx.x;
    if (tid >= total) return;
    int j    = tid & 7;
    int lane = (tid >> 3) & 31;
    int frag = tid >> 8;
    int COT = Cout >> 4, CT = Cin >> 5;
    int cot = frag % COT;
    int ct  = (frag / COT) % CT;
    int k   = frag / (COT * CT);
    int col  = cot * 16 + (lane & 15);
    int koff = (lane & 16) ? 8 : 0;
    int ci   = ct * 32 + koff + 2 * j + ((j >= 4) ? 8 : 0);
    unsigned lo = f2bf(W[((size_t)k * Cin + ci)     * Cout + col]);
    unsigned hi = f2bf(W[((size_t)k * Cin + ci + 1) * Cout + col]);
    Wp[(size_t)frag * 256 + lane * 8 + j] = lo | (hi << 16);
}

// ---------------------------------------------------------------------------
// Build X0 (N,192) bf16: scatter links through coords maps, gather 2D features.
// One block per voxel slot i, 192 threads = (v, c).
// ---------------------------------------------------------------------------
__global__ void gather2d_kernel(const float* __restrict__ feat2d,
                                const int* __restrict__ links,
                                const int* __restrict__ cmi,
                                const int* __restrict__ cmo,
                                unsigned short* __restrict__ X0) {
    int i = blockIdx.x;
    int t = threadIdx.x;            // 0..191
    int v = t >> 6;
    int c = t & 63;
    int src = cmi[i];
    int dst = cmo[i];
    const int* L = links + (size_t)src * 4 * V_NUM;   // (NL,4,V) layout
    int b   = L[0 * V_NUM + v];
    int h   = L[1 * V_NUM + v];
    int w   = L[2 * V_NUM + v];
    int val = L[3 * V_NUM + v];
    float f = feat2d[((((size_t)v * B_NUM + b) * C_CH + c) * H_IMG + h) * W_IMG + w];
    X0[(size_t)dst * (V_NUM * C_CH) + t] = f2bf(f * (float)val);
}

// feat3d_F (f32, N x 96) -> first 96 cols of CAT (bf16, stride 192)
__global__ void f3d_to_cat_kernel(const float* __restrict__ f3d,
                                  unsigned short* __restrict__ cat) {
    size_t e = (size_t)blockIdx.x * blockDim.x + threadIdx.x;
    if (e >= (size_t)N_VOX * D3_CH) return;
    size_t n = e / D3_CH;
    int    c = (int)(e % D3_CH);
    cat[n * (2 * D3_CH) + c] = f2bf(f3d[e]);
}

// ---------------------------------------------------------------------------
// Sparse gather-GEMM with bf16 WMMA.
// One wave computes a full 16 x COUT strip (16 voxels, all output channels):
// each gathered A fragment is loaded once and swept across all COT B
// fragments into COT accumulators -> 6x less gathered-A traffic vs. one wave
// per 16x16 tile. CIN/COUT are compile-time so all fragment loops unroll and
// tile decode needs no integer division.
// A fragment per ISA 16-bit 16x32 layout: lane<16 -> K {0..7,16..23},
// lane>=16 -> K {8..15,24..31}; two b128 loads per lane per 32-chunk.
// ---------------------------------------------------------------------------
template <int CIN, int COUT>
__global__ __launch_bounds__(128) void sconv_wmma_kernel(
    const unsigned short* __restrict__ X, const int* __restrict__ idx2,
    const uint32_t* __restrict__ Wp, float* __restrict__ Y, int rowTiles) {
    constexpr int CT  = CIN >> 5;    // K-chunks of 32
    constexpr int COT = COUT >> 4;   // column tiles of 16
    int lane = threadIdx.x & 31;
    int wave = threadIdx.x >> 5;
    int rt   = blockIdx.x * 4 + wave;
    if (rt >= rowTiles) return;
    int rbase  = rt << 4;
    int lrow   = lane & 15;
    int khalfA = (lane & 16) ? 8 : 0;

    v8f acc[COT];
#pragma unroll
    for (int c = 0; c < COT; ++c) acc[c] = {};

    for (int k = 0; k < K_NBR; ++k) {
        int  r     = idx2[(size_t)k * N_VOX + rbase + lrow];
        bool valid = (r >= 0);
        if (!__any(valid)) continue;                 // whole-strip tap skip
        const unsigned short* xr = X + (size_t)(valid ? r : 0) * CIN;
        const uint32_t*       wk = Wp + (size_t)k * (CT * COT * 256) + lane * 8;
#pragma unroll
        for (int ct = 0; ct < CT; ++ct) {
            Frag16 a;
            if (valid) {
                a.q[0] = *(const uint4*)(xr + ct * 32 + khalfA);
                a.q[1] = *(const uint4*)(xr + ct * 32 + 16 + khalfA);
            } else {
                a.q[0] = make_uint4(0u, 0u, 0u, 0u);
                a.q[1] = make_uint4(0u, 0u, 0u, 0u);
            }
#pragma unroll
            for (int cot = 0; cot < COT; ++cot) {
                Frag16 bfr;
                const uint4* wq = (const uint4*)(wk + ((size_t)ct * COT + cot) * 256);
                bfr.q[0] = wq[0];
                bfr.q[1] = wq[1];
                acc[cot] = __builtin_amdgcn_wmma_f32_16x16x32_bf16(
                    /*neg_a=*/false, a.v, /*neg_b=*/false, bfr.v,
                    /*c_mod=*/(short)0, acc[cot], /*reuse_a=*/false, /*reuse_b=*/false);
            }
        }
    }
    // C/D layout: VGPR j -> M = j (+8 for lanes 16..31), N = lane&15
    int rowoff = (lane & 16) ? 8 : 0;
#pragma unroll
    for (int cot = 0; cot < COT; ++cot) {
        float* yp = Y + (size_t)(rbase + rowoff) * COUT + cot * 16 + lrow;
#pragma unroll
        for (int j = 0; j < 8; ++j) yp[(size_t)j * COUT] = acc[cot][j];
    }
}

// ---------------------------------------------------------------------------
// Deterministic per-column sum / sumsq: one block per column, tree reduce.
// ---------------------------------------------------------------------------
__global__ void stats_kernel(const float* __restrict__ Y, int Cout,
                             float* __restrict__ stats) {
    __shared__ float ssum[256];
    __shared__ float ssq[256];
    int col = blockIdx.x;
    int tid = threadIdx.x;
    float s = 0.f, q = 0.f;
    for (int n = tid; n < N_VOX; n += 256) {
        float v = Y[(size_t)n * Cout + col];
        s += v; q += v * v;
    }
    ssum[tid] = s; ssq[tid] = q;
    __syncthreads();
    for (int st = 128; st > 0; st >>= 1) {
        if (tid < st) { ssum[tid] += ssum[tid + st]; ssq[tid] += ssq[tid + st]; }
        __syncthreads();
    }
    if (tid == 0) { stats[col] = ssum[0]; stats[Cout + col] = ssq[0]; }
}

// ---------------------------------------------------------------------------
// BN + ReLU. Writes bf16 (intermediate, with stride/offset for concat) or f32.
// ---------------------------------------------------------------------------
__global__ void bnrelu_kernel(const float* __restrict__ Y, int Cout,
                              const float* __restrict__ stats,
                              const float* __restrict__ g, const float* __restrict__ b,
                              unsigned short* __restrict__ outb, float* __restrict__ outf,
                              int ostride, int ooff) {
    size_t e = (size_t)blockIdx.x * blockDim.x + threadIdx.x;
    if (e >= (size_t)N_VOX * Cout) return;
    size_t n = e / Cout;
    int    c = (int)(e % Cout);
    float inv_n = 1.f / (float)N_VOX;
    float mu  = stats[c] * inv_n;
    float var = stats[Cout + c] * inv_n - mu * mu;
    float y = g[c] * (Y[e] - mu) * rsqrtf(var + EPSV) + b[c];
    y = y > 0.f ? y : 0.f;
    if (outf) outf[n * (size_t)ostride + ooff + c] = y;
    else      outb[n * (size_t)ostride + ooff + c] = f2bf(y);
}

// ---------------------------------------------------------------------------
extern "C" void kernel_launch(void* const* d_in, const int* in_sizes, int n_in,
                              void* d_out, int out_size, void* d_ws, size_t ws_size,
                              hipStream_t stream) {
    (void)in_sizes; (void)n_in; (void)out_size; (void)ws_size;
    const float* feat2d = (const float*)d_in[0];
    const float* f3d    = (const float*)d_in[1];
    const int*   links  = (const int*)d_in[2];
    const int*   cmi    = (const int*)d_in[3];
    const int*   cmo    = (const int*)d_in[4];
    const int*   in_idx = (const int*)d_in[5];
    const unsigned char* mask = (const unsigned char*)d_in[6];   // jnp bool = 1 byte
    const float* W1 = (const float*)d_in[7];
    const float* g1 = (const float*)d_in[8];
    const float* b1 = (const float*)d_in[9];
    const float* W2 = (const float*)d_in[10];
    const float* g2 = (const float*)d_in[11];
    const float* b2 = (const float*)d_in[12];
    const float* W3 = (const float*)d_in[13];
    const float* g3 = (const float*)d_in[14];
    const float* b3 = (const float*)d_in[15];

    // Workspace carve-out (256B aligned slabs)
    char* ws = (char*)d_ws;
    size_t off = 0;
    auto alloc = [&](size_t bytes) -> char* {
        char* p = ws + off;
        off = (off + bytes + 255) & ~(size_t)255;
        return p;
    };
    int*            idx2  = (int*)alloc((size_t)K_NBR * N_VOX * 4);
    uint32_t*       wp1   = (uint32_t*)alloc((size_t)K_NBR * 6 * 4 * 256 * 4);  // Cin 192, Cout 64
    uint32_t*       wp2   = (uint32_t*)alloc((size_t)K_NBR * 2 * 6 * 256 * 4);  // Cin 64,  Cout 96
    uint32_t*       wp3   = (uint32_t*)alloc((size_t)K_NBR * 6 * 6 * 256 * 4);  // Cin 192, Cout 96
    unsigned short* X0    = (unsigned short*)alloc((size_t)N_VOX * 192 * 2);
    unsigned short* H1    = (unsigned short*)alloc((size_t)N_VOX * 64 * 2);
    unsigned short* CAT   = (unsigned short*)alloc((size_t)N_VOX * 192 * 2);
    float*          Y     = (float*)alloc((size_t)N_VOX * 96 * 4);
    float*          stats = (float*)alloc(3 * 192 * 4);

    const int rowTiles = N_VOX / 16;                 // 12500 (exact)
    const int sconvBlocks = (rowTiles + 3) / 4;      // 4 waves per block

    // Preparation
    { int tot = K_NBR * N_VOX;
      prep_idx_kernel<<<(tot + 255) / 256, 256, 0, stream>>>(in_idx, mask, idx2, tot); }
    { int tot = K_NBR * 6 * 4 * 256;
      pack_w_kernel<<<(tot + 255) / 256, 256, 0, stream>>>(W1, wp1, 192, 64, tot); }
    { int tot = K_NBR * 2 * 6 * 256;
      pack_w_kernel<<<(tot + 255) / 256, 256, 0, stream>>>(W2, wp2, 64, 96, tot); }
    { int tot = K_NBR * 6 * 6 * 256;
      pack_w_kernel<<<(tot + 255) / 256, 256, 0, stream>>>(W3, wp3, 192, 96, tot); }
    gather2d_kernel<<<N_VOX, 192, 0, stream>>>(feat2d, links, cmi, cmo, X0);
    { size_t tot = (size_t)N_VOX * D3_CH;
      f3d_to_cat_kernel<<<(int)((tot + 255) / 256), 256, 0, stream>>>(f3d, CAT); }

    // Layer 1: (N,192) -> (N,64)
    sconv_wmma_kernel<192, 64><<<sconvBlocks, 128, 0, stream>>>(X0, idx2, wp1, Y, rowTiles);
    stats_kernel<<<64, 256, 0, stream>>>(Y, 64, stats);
    { size_t tot = (size_t)N_VOX * 64;
      bnrelu_kernel<<<(int)((tot + 255) / 256), 256, 0, stream>>>(Y, 64, stats, g1, b1, H1, nullptr, 64, 0); }

    // Layer 2: (N,64) -> (N,96), BN output goes into CAT columns 96..191
    sconv_wmma_kernel<64, 96><<<sconvBlocks, 128, 0, stream>>>(H1, idx2, wp2, Y, rowTiles);
    stats_kernel<<<96, 256, 0, stream>>>(Y, 96, stats + 192);
    { size_t tot = (size_t)N_VOX * 96;
      bnrelu_kernel<<<(int)((tot + 255) / 256), 256, 0, stream>>>(Y, 96, stats + 192, g2, b2, CAT, nullptr, 192, 96); }

    // Layer 3: (N,192) -> (N,96), BN output -> d_out (f32)
    sconv_wmma_kernel<192, 96><<<sconvBlocks, 128, 0, stream>>>(CAT, idx2, wp3, Y, rowTiles);
    stats_kernel<<<96, 256, 0, stream>>>(Y, 96, stats + 384);
    { size_t tot = (size_t)N_VOX * 96;
      bnrelu_kernel<<<(int)((tot + 255) / 256), 256, 0, stream>>>(Y, 96, stats + 384, g3, b3, nullptr, (float*)d_out, 96, 0); }
}
